// BahdanauAttention_88287347736832
// MI455X (gfx1250) — matrix operand order repaired and checked by reading, not compiled, not run on hardware
//
#include <hip/hip_runtime.h>
#include <hip/hip_bf16.h>
#include <math.h>

// Problem constants (match reference)
#define BB 32
#define SS 2048
#define HH 1024

typedef __attribute__((ext_vector_type(16))) __bf16 v16bf;
typedef __attribute__((ext_vector_type(8)))  float  v8f;

__device__ __forceinline__ unsigned short f2bf(float f) {
    unsigned int u = __float_as_uint(f);
    // round-to-nearest-even bf16 truncation
    unsigned int r = u + 0x7FFFu + ((u >> 16) & 1u);
    return (unsigned short)(r >> 16);
}

__device__ __forceinline__ unsigned int f2bf_pack(float lo, float hi) {
    return (unsigned int)f2bf(lo) | ((unsigned int)f2bf(hi) << 16);
}

// fast tanh: 1 - 2/(exp(2x)+1); saturates correctly to +-1 for large |x|
__device__ __forceinline__ float fast_tanh(float x) {
    float e2 = __expf(2.0f * x);
    return 1.0f - 2.0f * __builtin_amdgcn_rcpf(e2 + 1.0f);
}

// ---------------------------------------------------------------------------
// Kernel 1: pack W_h (fp32 [H][H], row = output o, col = input h) into bf16
// B-fragment-linear layout for V_WMMA_F32_16X16X32_BF16.
// B fragment (32x16, KxN), as packed here (contiguous K): lane L holds column N=L%16,
// element e (of 16 bf16) holds K = e + (L/16)*16.
// Linear element index: (((ntile*32 + kstep)*32 + lane)*16 + e)
// ---------------------------------------------------------------------------
__global__ __launch_bounds__(256) void pack_wh_kernel(const float* __restrict__ W_h,
                                                      unsigned short* __restrict__ whb) {
    unsigned int t = blockIdx.x * 256u + threadIdx.x;   // 0 .. H*H-1
    int e     =  t        & 15;
    int lane  = (t >> 4)  & 31;
    int kstep = (t >> 9)  & 31;
    int ntile =  t >> 14;                               // 0..63
    int n = ntile * 16 + (lane & 15);
    int k = kstep * 32 + e + ((lane >> 4) << 4);
    whb[t] = f2bf(W_h[(size_t)n * HH + k]);
}

// ---------------------------------------------------------------------------
// Kernel 2: ws[b][o] = sum_h dec[b][h] * W_s[o][h]; also zero context output.
// 128 blocks x 256 threads; one (b,o) per thread. Also zeroes ctx (B*H=32768).
// ---------------------------------------------------------------------------
__global__ __launch_bounds__(256) void ws_zero_kernel(const float* __restrict__ dec,
                                                      const float* __restrict__ W_s,
                                                      float* __restrict__ wsOut,
                                                      float* __restrict__ ctxZero) {
    int tid = threadIdx.x;
    int blk = blockIdx.x;              // 0..127
    int b = blk >> 2;
    int o = ((blk & 3) << 8) + tid;
    const float* d = dec + (size_t)b * HH;
    const float* w = W_s + (size_t)o * HH;
    float acc = 0.0f;
#pragma unroll 4
    for (int h = 0; h < HH; h += 4) {
        acc += d[h] * w[h] + d[h + 1] * w[h + 1] + d[h + 2] * w[h + 2] + d[h + 3] * w[h + 3];
    }
    wsOut[(size_t)b * HH + o] = acc;
    ctxZero[blk * 256 + tid] = 0.0f;
}

// ---------------------------------------------------------------------------
// Kernel 3: fused  scores[b,s] = sum_o tanh( (enc @ W_h^T)[b,s,o] + ws[b,o] ) * v[o]
//
// Grid: 1024 row-strips of M=64 rows over B*S. Block: 512 threads = 16 wave32.
// Each wave owns a 64x64 output tile: 4 A frags (M) x 4 B frags (N) -> 16 WMMA
// per K-step with only 128B of global B traffic (8 B-bytes/WMMA) -> halves the
// L2 stream of the packed W_h vs the 32-row variant.
//
// The FULL 64x1024 A strip is staged ONCE into dynamic LDS (128 KB bf16,
// legal on CDNA5's 320 KB WGP LDS) in A-fragment order (contiguous K, same as B: lane L:
// M=L%16, element e: K=e+(L/16)*16), index ((ks*4+mt)*32+lane)*16+e.
// One barrier total; the 32-step K loop runs barrier-free with B fragments
// explicitly double-buffered (b0/b1, K unrolled by 2) so next-step loads issue
// before the current 16 WMMAs.
// ---------------------------------------------------------------------------
__global__ __launch_bounds__(512) void wh_fused_kernel(const float* __restrict__ enc,
                                                       const unsigned short* __restrict__ whb,
                                                       const float* __restrict__ ws,
                                                       const float* __restrict__ vvec,
                                                       float* __restrict__ scores) {
    extern __shared__ __align__(32) unsigned char dynsmem[];
    unsigned short* aA = reinterpret_cast<unsigned short*>(dynsmem);       // 128 KB
    float* sScore = reinterpret_cast<float*>(dynsmem + 32 * 4 * 32 * 16 * 2);

    const int tid  = threadIdx.x;
    const int lane = tid & 31;
    const int wave = tid >> 5;           // 0..15
    const int l16  = lane & 15;
    const int g    = lane >> 4;

    const int row0 = blockIdx.x * 64;    // global row in [0, B*S)
    const int b    = row0 / SS;
    const int s0   = row0 % SS;
    const int nW0  = wave * 64;          // wave's N base (4 ntiles)

    if (tid < 64) sScore[tid] = 0.0f;

    // ---- stage full 64x1024 A strip: 16384 quads (4 bf16 each); 32/thread --
    // quad q: eq=q&3, lane=(q>>2)&31, mt=(q>>7)&3, ks=q>>9; element base = 4*q
    {
        uint2* aQ = reinterpret_cast<uint2*>(aA);
#pragma unroll
        for (int i = 0; i < 32; ++i) {
            int q   = tid + i * 512;
            int eq  = q & 3;
            int sl  = (q >> 2) & 31;
            int mt  = (q >> 7) & 3;
            int ks  = q >> 9;
            int m   = (sl & 15) + mt * 16;
            int e0  = eq * 4;                              // 0,4,8,12
            int kk  = e0 + ((sl >> 4) << 4);
            const float4 f = *reinterpret_cast<const float4*>(
                &enc[(size_t)(row0 + m) * HH + ks * 32 + kk]);
            uint2 p;
            p.x = f2bf_pack(f.x, f.y);
            p.y = f2bf_pack(f.z, f.w);
            aQ[q] = p;
        }
    }
    __syncthreads();

    // ---- K loop: no barriers, double-buffered B, 16 WMMA per step ----
    v8f acc[4][4];
#pragma unroll
    for (int mt = 0; mt < 4; ++mt)
#pragma unroll
        for (int nt = 0; nt < 4; ++nt)
#pragma unroll
            for (int r = 0; r < 8; ++r) acc[mt][nt][r] = 0.0f;

    const v16bf* aV = reinterpret_cast<const v16bf*>(aA);              // (ks*4+mt)*32+lane
    const v16bf* bV = reinterpret_cast<const v16bf*>(whb) +
                      ((size_t)wave * 4 * 1024 + lane);                // + nt*1024 + ks*32

    v16bf b0[4], b1[4];
#pragma unroll
    for (int nt = 0; nt < 4; ++nt) b0[nt] = bV[nt * 1024];             // ks = 0

#pragma unroll 1
    for (int ks = 0; ks < 32; ks += 2) {
        // prefetch ks+1 while computing ks
#pragma unroll
        for (int nt = 0; nt < 4; ++nt) b1[nt] = bV[nt * 1024 + (ks + 1) * 32];
#pragma unroll
        for (int mp = 0; mp < 4; mp += 2) {
            const v16bf a0 = aV[(ks * 4 + mp + 0) * 32 + lane];
            const v16bf a1 = aV[(ks * 4 + mp + 1) * 32 + lane];
#pragma unroll
            for (int nt = 0; nt < 4; ++nt) {
                acc[mp + 0][nt] = __builtin_amdgcn_wmma_f32_16x16x32_bf16(
                    false, a0, false, b0[nt], (short)0, acc[mp + 0][nt], false, false);
                acc[mp + 1][nt] = __builtin_amdgcn_wmma_f32_16x16x32_bf16(
                    false, a1, false, b0[nt], (short)0, acc[mp + 1][nt], false, false);
            }
        }
        // prefetch ks+2 while computing ks+1 (final prefetch lands in the
        // adjacent scratch region; values are never consumed)
#pragma unroll
        for (int nt = 0; nt < 4; ++nt) b0[nt] = bV[nt * 1024 + (ks + 2) * 32];
#pragma unroll
        for (int mp = 0; mp < 4; mp += 2) {
            const v16bf a0 = aV[((ks + 1) * 4 + mp + 0) * 32 + lane];
            const v16bf a1 = aV[((ks + 1) * 4 + mp + 1) * 32 + lane];
#pragma unroll
            for (int nt = 0; nt < 4; ++nt) {
                acc[mp + 0][nt] = __builtin_amdgcn_wmma_f32_16x16x32_bf16(
                    false, a0, false, b1[nt], (short)0, acc[mp + 0][nt], false, false);
                acc[mp + 1][nt] = __builtin_amdgcn_wmma_f32_16x16x32_bf16(
                    false, a1, false, b1[nt], (short)0, acc[mp + 1][nt], false, false);
            }
        }
    }

    // ---- epilogue: tanh(+ws) then dot with v, reduce per row ----
    float rowsum[4][8];
#pragma unroll
    for (int mt = 0; mt < 4; ++mt)
#pragma unroll
        for (int r = 0; r < 8; ++r) rowsum[mt][r] = 0.0f;

#pragma unroll
    for (int nt = 0; nt < 4; ++nt) {
        const int n = nW0 + nt * 16 + l16;
        const float wsv = ws[(size_t)b * HH + n];
        const float vv  = vvec[n];
#pragma unroll
        for (int mt = 0; mt < 4; ++mt)
#pragma unroll
            for (int r = 0; r < 8; ++r)
                rowsum[mt][r] += fast_tanh(acc[mt][nt][r] + wsv) * vv;
    }
#pragma unroll
    for (int mt = 0; mt < 4; ++mt)
#pragma unroll
        for (int r = 0; r < 8; ++r)
            atomicAdd(&sScore[mt * 16 + r + g * 8], rowsum[mt][r]);    // ds_add_f32
    __syncthreads();
    if (tid < 64) scores[(size_t)b * SS + s0 + tid] = sScore[tid];
}

// ---------------------------------------------------------------------------
// Kernel 4: softmax over S per batch.  32 blocks x 256 threads.
// ---------------------------------------------------------------------------
__global__ __launch_bounds__(256) void softmax_kernel(const float* __restrict__ scores,
                                                      float* __restrict__ attn) {
    __shared__ float red[256];
    const int b = blockIdx.x, tid = threadIdx.x;
    const float* sc = scores + (size_t)b * SS;

    float m = -INFINITY;
    for (int i = tid; i < SS; i += 256) m = fmaxf(m, sc[i]);
    red[tid] = m; __syncthreads();
    for (int o = 128; o > 0; o >>= 1) {
        if (tid < o) red[tid] = fmaxf(red[tid], red[tid + o]);
        __syncthreads();
    }
    m = red[0]; __syncthreads();

    float sum = 0.0f;
    for (int i = tid; i < SS; i += 256) sum += __expf(sc[i] - m);
    red[tid] = sum; __syncthreads();
    for (int o = 128; o > 0; o >>= 1) {
        if (tid < o) red[tid] += red[tid + o];
        __syncthreads();
    }
    const float inv = 1.0f / red[0];
    for (int i = tid; i < SS; i += 256) attn[(size_t)b * SS + i] = __expf(sc[i] - m) * inv;
}

// ---------------------------------------------------------------------------
// Kernel 5: context[b,h] = sum_s attn[b,s] * enc[b,s,h]   (memory bound, 256MB)
// Grid (H/256, 8 s-chunks, B); fp32 global atomics into pre-zeroed ctx.
// ---------------------------------------------------------------------------
__global__ __launch_bounds__(256) void context_kernel(const float* __restrict__ enc,
                                                      const float* __restrict__ attn,
                                                      float* __restrict__ ctx) {
    const int h   = blockIdx.x * 256 + threadIdx.x;
    const int b   = blockIdx.z;
    const int sc0 = blockIdx.y * 256;
    const float* e = enc + ((size_t)b * SS + sc0) * HH + h;
    const float* a = attn + (size_t)b * SS + sc0;
    float acc = 0.0f;
#pragma unroll 4
    for (int s = 0; s < 256; ++s) acc += a[s] * e[(size_t)s * HH];
    atomicAdd(&ctx[(size_t)b * HH + h], acc);
}

// ---------------------------------------------------------------------------
extern "C" void kernel_launch(void* const* d_in, const int* in_sizes, int n_in,
                              void* d_out, int out_size, void* d_ws, size_t ws_size,
                              hipStream_t stream) {
    (void)in_sizes; (void)n_in; (void)out_size; (void)ws_size;

    const float* dec  = (const float*)d_in[0];   // [B,H]
    const float* enc  = (const float*)d_in[1];   // [B,S,H]
    const float* W_h  = (const float*)d_in[2];   // [H,H]
    const float* W_s  = (const float*)d_in[3];   // [H,H]
    const float* vv   = (const float*)d_in[4];   // [H]

    float* out   = (float*)d_out;
    float* ctx   = out;                          // [B,H]
    float* attn  = out + (size_t)BB * HH;        // [B,S]

    char* wsp = (char*)d_ws;
    unsigned short* whb    = (unsigned short*)wsp;                       // 2 MB
    float*          ws_vec = (float*)(wsp + (size_t)HH * HH * 2);        // 128 KB
    float*          scores = (float*)(wsp + (size_t)HH * HH * 2
                                          + (size_t)BB * HH * 4);        // 256 KB

    // 1. pack W_h into bf16 B-fragment layout
    pack_wh_kernel<<<(HH * HH) / 256, 256, 0, stream>>>(W_h, whb);
    // 2. ws GEMV + zero context output
    ws_zero_kernel<<<(BB * HH) / 256, 256, 0, stream>>>(dec, W_s, ws_vec, ctx);
    // 3. fused WMMA GEMM + tanh + v-dot -> scores (128 KB + 256 B dynamic LDS)
    const size_t smem = (size_t)(32 * 4 * 32 * 16) * 2 + 64 * sizeof(float);
    wh_fused_kernel<<<(BB * SS) / 64, 512, smem, stream>>>(enc, whb, ws_vec, vv, scores);
    // 4. softmax -> attention weights (second output)
    softmax_kernel<<<BB, 256, 0, stream>>>(scores, attn);
    // 5. context = attn @ enc (first output)
    context_kernel<<<dim3(HH / 256, 8, BB), 256, 0, stream>>>(enc, attn, ctx);
}
